// FlaxLlamaAttention_29240137351508
// MI455X (gfx1250) — matrix operand order repaired and checked by reading, not compiled, not run on hardware
//
#include <hip/hip_runtime.h>
#include <stdint.h>

#define S_LEN 2048
#define HDIM  4096
#define NHq   32
#define NKVh  8
#define HD    128
#define REPS  4

typedef __attribute__((ext_vector_type(16))) __bf16 v16bf;
typedef __attribute__((ext_vector_type(8)))  float  v8f;
typedef __attribute__((ext_vector_type(4)))  int    aint4;

#define ASG __attribute__((address_space(1)))
#define ASL __attribute__((address_space(3)))

union Frag {
    uint4 u[2];
    unsigned int w[8];
    v16bf v;
};

#if defined(__HIP_DEVICE_COMPILE__) && \
    __has_builtin(__builtin_amdgcn_global_load_async_to_lds_b128) && \
    __has_builtin(__builtin_amdgcn_s_wait_asynccnt)
#define USE_ASYNC_LDS 1
#else
#define USE_ASYNC_LDS 0
#endif

__device__ __forceinline__ unsigned short bf16_rn(float f) {
    unsigned int u = __float_as_uint(f);
    unsigned int r = (u + 0x7FFFu + ((u >> 16) & 1u)) >> 16;
    return (unsigned short)r;
}
__device__ __forceinline__ unsigned int pack2_bf16(float lo, float hi) {
    return (unsigned int)bf16_rn(lo) | ((unsigned int)bf16_rn(hi) << 16);
}
__device__ __forceinline__ v8f wmma_bf16(v16bf a, v16bf b, v8f c) {
    return __builtin_amdgcn_wmma_f32_16x16x32_bf16(false, a, false, b, (short)0, c, false, false);
}

// Stage 16B from global to LDS (async if available).
__device__ __forceinline__ void stage16(const unsigned short* g, unsigned short* l) {
#if USE_ASYNC_LDS
    __builtin_amdgcn_global_load_async_to_lds_b128(
        (ASG aint4*)(void*)g, (ASL aint4*)(void*)l, 0, 0);
#else
    *(uint4*)l = *(const uint4*)g;
#endif
}
__device__ __forceinline__ void stage_wait() {
#if USE_ASYNC_LDS
    __builtin_amdgcn_s_wait_asynccnt(0);
#endif
}

// ---------------- f32 -> bf16 convert (row major kept) ----------------
__global__ void k_f32_to_bf16(const float* __restrict__ src,
                              unsigned short* __restrict__ dst, int n) {
    int i = blockIdx.x * blockDim.x + threadIdx.x;
    int stride = gridDim.x * blockDim.x;
    for (; i < n; i += stride) dst[i] = bf16_rn(src[i]);
}

// ------- f32 [K,N] -> bf16 transposed [N,K] (tiled, coalesced both ways) ----
__global__ __launch_bounds__(256) void k_transpose_f32_bf16(
    const float* __restrict__ src, unsigned short* __restrict__ dst, int K, int N) {
    __shared__ float tile[32][33];
    const int tx = threadIdx.x, ty = threadIdx.y;  // 32 x 8
    const int kb = blockIdx.y * 32, nb = blockIdx.x * 32;
    #pragma unroll
    for (int j = 0; j < 32; j += 8)
        tile[ty + j][tx] = src[(size_t)(kb + ty + j) * N + nb + tx];
    __syncthreads();
    #pragma unroll
    for (int j = 0; j < 32; j += 8)
        dst[(size_t)(nb + ty + j) * K + kb + tx] = bf16_rn(tile[tx][ty + j]);
}

// -------- bf16 GEMM: C[M,N] = A[M,K] * Bt[N,K]^T, C f32 --------------------
// block = 256 thr (8 waves), block tile 128x256, wave tile 64x64, K step 32.
// Double-buffered LDS tiles, staged with async b128 copies: one barrier per
// K-step, copy of tile i+1 fully overlapped with the 16 WMMAs of tile i.
__global__ __launch_bounds__(256) void k_gemm_bf16(
    const unsigned short* __restrict__ A, const unsigned short* __restrict__ Bt,
    float* __restrict__ C, int M, int N, int K) {
    __shared__ __align__(16) unsigned short lA[2][128 * 32];
    __shared__ __align__(16) unsigned short lB[2][256 * 32];
    const int t = threadIdx.x;
    const int lane = t & 31;
    const int wave = t >> 5;
    const int n = lane & 15;
    const int hlf = lane >> 4;
    const int wm = (wave & 1) * 64;
    const int wn = (wave >> 1) * 64;
    const int m0 = blockIdx.y * 128, n0 = blockIdx.x * 256;

    auto stage_tiles = [&](int buf, int k0) {
        // A tile [128 x 32]: 4096 elems = 256 thr * 2 * 8
        #pragma unroll
        for (int it = 0; it < 2; ++it) {
            int e = it * 2048 + t * 8;
            stage16(A + (size_t)(m0 + (e >> 5)) * K + k0 + (e & 31), &lA[buf][e]);
        }
        // Bt tile [256 x 32]: 8192 elems = 256 thr * 4 * 8
        #pragma unroll
        for (int it = 0; it < 4; ++it) {
            int e = it * 2048 + t * 8;
            stage16(Bt + (size_t)(n0 + (e >> 5)) * K + k0 + (e & 31), &lB[buf][e]);
        }
    };

    v8f acc[4][4] = {};
    stage_tiles(0, 0);

    for (int k0 = 0; k0 < K; k0 += 32) {
        const int buf = (k0 >> 5) & 1;
        stage_wait();      // tile 'buf' copy (issued last iteration) complete
        __syncthreads();   // all waves: reads of other buffer done, writes visible
        if (k0 + 32 < K) stage_tiles(buf ^ 1, k0 + 32);  // prefetch next tile

        Frag a[4], b[4];
        #pragma unroll
        for (int mt = 0; mt < 4; ++mt) {
            // A frag layout: h0 j0-3:K0-7 j4-7:K16-23, h1 j0-3:K8-15 j4-7:K24-31
            const char* base = (const char*)(&lA[buf][(wm + mt * 16 + n) * 32]);
            a[mt].u[0] = *(const uint4*)(base + hlf * 16);
            a[mt].u[1] = *(const uint4*)(base + 32 + hlf * 16);
        }
        #pragma unroll
        for (int nt = 0; nt < 4; ++nt) {
            // B frag layout: h0 j0-3:K0-7 j4-7:K16-23, h1 j0-3:K8-15 j4-7:K24-31
            const char* base = (const char*)(&lB[buf][(wn + nt * 16 + n) * 32]);
            b[nt].u[0] = *(const uint4*)(base + hlf * 16);
            b[nt].u[1] = *(const uint4*)(base + 32 + hlf * 16);
        }
        #pragma unroll
        for (int mt = 0; mt < 4; ++mt)
            #pragma unroll
            for (int nt = 0; nt < 4; ++nt)
                acc[mt][nt] = wmma_bf16(a[mt].v, b[nt].v, acc[mt][nt]);
    }
    // C layout: lane n = col, VGPR r -> row r + 8*half
    #pragma unroll
    for (int mt = 0; mt < 4; ++mt)
        #pragma unroll
        for (int nt = 0; nt < 4; ++nt) {
            const int col = n0 + wn + nt * 16 + n;
            const int row0 = m0 + wm + mt * 16 + 8 * hlf;
            #pragma unroll
            for (int r = 0; r < 8; ++r)
                C[(size_t)(row0 + r) * N + col] = acc[mt][nt][r];
        }
}

// ---------------- RoPE: src [S, nh*HD] f32 -> dst [nh, S, HD] bf16 ----------
__global__ void k_rope(const float* __restrict__ src, unsigned short* __restrict__ dst,
                       const float* __restrict__ sinT, const float* __restrict__ cosT,
                       const int* __restrict__ pos_ids, int nheads) {
    int idx = blockIdx.x * blockDim.x + threadIdx.x;
    int total = S_LEN * nheads * HD;
    if (idx >= total) return;
    int d = idx & (HD - 1);
    int h = (idx >> 7) % nheads;
    int s = idx / (HD * nheads);
    int pos = pos_ids[s];
    int base = s * nheads * HD + h * HD;
    float x = src[base + d];
    float other = (d < 64) ? -src[base + d + 64] : src[base + d - 64];
    float r = x * cosT[pos * HD + d] + other * sinT[pos * HD + d];
    dst[((size_t)h * S_LEN + s) * HD + d] = bf16_rn(r);
}

// ---------------- V transpose: [S, NKV*HD] f32 -> [NKV*HD, S] bf16 ----------
__global__ void k_vtrans(const float* __restrict__ src, unsigned short* __restrict__ dst) {
    int idx = blockIdx.x * blockDim.x + threadIdx.x;
    if (idx >= S_LEN * NKVh * HD) return;
    int s = idx & (S_LEN - 1);
    int hd = idx >> 11;  // h*128 + d
    float v = src[(size_t)s * (NKVh * HD) + hd];
    dst[(size_t)hd * S_LEN + s] = bf16_rn(v);
}

// ---------------- Flash attention (per wave: 16 queries, one head) ----------
// S^T = K * Q^T and O^T = V^T * P^T keep softmax and the P^T fragment build
// lane-local (one shfl_xor(16) per reduction / exchange).
__global__ __launch_bounds__(256) void k_attention(
    const unsigned short* __restrict__ qh,  // [NH, S, HD] bf16 (post-RoPE)
    const unsigned short* __restrict__ kh,  // [NKV, S, HD] bf16 (post-RoPE)
    const unsigned short* __restrict__ vT,  // [NKV, HD, S] bf16
    const int* __restrict__ amask,          // [S]
    unsigned short* __restrict__ attn) {    // [S, NH*HD] bf16
    const int lane = threadIdx.x & 31;
    const int wave = threadIdx.x >> 5;
    const int n = lane & 15;
    const int hlf = lane >> 4;
    const int head = blockIdx.x;
    const int q0 = blockIdx.y * 128 + wave * 16;
    const int kvh = head / REPS;
    const float scale = 0.08838834764831845f;  // 1/sqrt(128)

    Frag qf[4];
    const unsigned short* qrow = qh + ((size_t)head * S_LEN + q0 + n) * HD;
    #pragma unroll
    for (int c = 0; c < 4; ++c) {
        const char* p = (const char*)(qrow + c * 32);
        qf[c].u[0] = *(const uint4*)(p + hlf * 16);
        qf[c].u[1] = *(const uint4*)(p + 32 + hlf * 16);
    }

    v8f acc[8] = {};
    float m_i = -3.0e38f, l_i = 0.0f;
    const int qg = q0 + n;
    const int kend = q0 + 16;

    for (int kb = 0; kb < kend; kb += 32) {
        v8f st[2] = {};
        #pragma unroll
        for (int tt = 0; tt < 2; ++tt) {
            const unsigned short* krow =
                kh + ((size_t)kvh * S_LEN + kb + tt * 16 + n) * HD;
            #pragma unroll
            for (int c = 0; c < 4; ++c) {
                Frag kf;
                const char* p = (const char*)(krow + c * 32);
                kf.u[0] = *(const uint4*)(p + hlf * 16);
                kf.u[1] = *(const uint4*)(p + 32 + hlf * 16);
                st[tt] = wmma_bf16(kf.v, qf[c].v, st[tt]);
            }
        }
        int am = amask[kb + lane];
        float vals[2][8];
        float bmax = -3.0e38f;
        #pragma unroll
        for (int tt = 0; tt < 2; ++tt)
            #pragma unroll
            for (int r = 0; r < 8; ++r) {
                int krel = tt * 16 + r + 8 * hlf;
                int kg = kb + krel;
                int amk = __shfl(am, krel, 32);
                float v = st[tt][r] * scale;
                if (kg > qg || amk == 0) v = -3.0e38f;
                vals[tt][r] = v;
                bmax = fmaxf(bmax, v);
            }
        bmax = fmaxf(bmax, __shfl_xor(bmax, 16, 32));
        float mnew = fmaxf(m_i, bmax);
        float corr = __expf(m_i - mnew);
        float bsum = 0.0f;
        unsigned int pk[2][4];
        #pragma unroll
        for (int tt = 0; tt < 2; ++tt) {
            float e[8];
            #pragma unroll
            for (int r = 0; r < 8; ++r) {
                e[r] = __expf(vals[tt][r] - mnew);
                bsum += e[r];
            }
            #pragma unroll
            for (int i = 0; i < 4; ++i) pk[tt][i] = pack2_bf16(e[2 * i], e[2 * i + 1]);
        }
        bsum += __shfl_xor(bsum, 16, 32);
        l_i = l_i * corr + bsum;
        m_i = mnew;
        #pragma unroll
        for (int dt = 0; dt < 8; ++dt)
            #pragma unroll
            for (int r = 0; r < 8; ++r) acc[dt][r] *= corr;

        Frag pb;
        #pragma unroll
        for (int i = 0; i < 4; ++i) {
            unsigned int r0 = __shfl_xor(pk[0][i], 16, 32);
            unsigned int r1 = __shfl_xor(pk[1][i], 16, 32);
            pb.w[i]     = hlf ? r1 : pk[0][i];
            pb.w[4 + i] = hlf ? pk[1][i] : r0;
        }
        #pragma unroll
        for (int dt = 0; dt < 8; ++dt) {
            Frag vf;
            const char* p =
                (const char*)(vT + ((size_t)kvh * HD + dt * 16 + n) * S_LEN + kb);
            vf.u[0] = *(const uint4*)(p + hlf * 32);
            vf.u[1] = *(const uint4*)(p + hlf * 32 + 16);
            acc[dt] = wmma_bf16(vf.v, pb.v, acc[dt]);
        }
    }
    float invl = 1.0f / l_i;
    #pragma unroll
    for (int dt = 0; dt < 8; ++dt)
        #pragma unroll
        for (int r = 0; r < 8; ++r) {
            int hd = dt * 16 + r + 8 * hlf;
            attn[(size_t)(q0 + n) * (NHq * HD) + head * HD + hd] =
                bf16_rn(acc[dt][r] * invl);
        }
}

// ---------------- launch ----------------
extern "C" void kernel_launch(void* const* d_in, const int* in_sizes, int n_in,
                              void* d_out, int out_size, void* d_ws, size_t ws_size,
                              hipStream_t stream) {
    const float* hidden = (const float*)d_in[0];
    const float* wq = (const float*)d_in[1];
    const float* wk = (const float*)d_in[2];
    const float* wv = (const float*)d_in[3];
    const float* wo = (const float*)d_in[4];
    const float* sinT = (const float*)d_in[5];
    const float* cosT = (const float*)d_in[6];
    const int* amask = (const int*)d_in[7];
    const int* pos = (const int*)d_in[8];
    float* out = (float*)d_out;

    char* ws = (char*)d_ws;
    size_t off = 0;
    auto alloc = [&](size_t bytes) {
        char* p = ws + off;
        off += (bytes + 255) & ~(size_t)255;
        return p;
    };
    unsigned short* wq_t  = (unsigned short*)alloc((size_t)4096 * 4096 * 2);  // [N,K]
    unsigned short* wk_t  = (unsigned short*)alloc((size_t)1024 * 4096 * 2);
    unsigned short* wv_t  = (unsigned short*)alloc((size_t)1024 * 4096 * 2);
    unsigned short* wo_t  = (unsigned short*)alloc((size_t)4096 * 4096 * 2);
    unsigned short* hid_b = (unsigned short*)alloc((size_t)2048 * 4096 * 2);
    float* q_f = (float*)alloc((size_t)2048 * 4096 * 4);
    float* k_f = (float*)alloc((size_t)2048 * 1024 * 4);
    float* v_f = (float*)alloc((size_t)2048 * 1024 * 4);
    // Aliases: target regions are dead by the time these are written.
    unsigned short* attn_b = wq_t;   // 16MB into 32MB (wq dead after Q GEMM)
    unsigned short* k_b    = wk_t;   //  4MB into  8MB
    unsigned short* vT_b   = wv_t;   //  4MB into  8MB
    unsigned short* q_b    = hid_b;  // 16MB into 16MB (hidden dead)

    // 1) converts (weights transposed to [N,K] so GEMM staging is b128-only)
    dim3 tb(32, 8);
    k_f32_to_bf16<<<4096, 256, 0, stream>>>(hidden, hid_b, 2048 * 4096);
    k_transpose_f32_bf16<<<dim3(128, 128), tb, 0, stream>>>(wq, wq_t, 4096, 4096);
    k_transpose_f32_bf16<<<dim3(32, 128), tb, 0, stream>>>(wk, wk_t, 4096, 1024);
    k_transpose_f32_bf16<<<dim3(32, 128), tb, 0, stream>>>(wv, wv_t, 4096, 1024);
    k_transpose_f32_bf16<<<dim3(128, 128), tb, 0, stream>>>(wo, wo_t, 4096, 4096);
    // 2) QKV projections (block tile 128x256)
    dim3 gq(4096 / 256, 2048 / 128), gkv(1024 / 256, 2048 / 128);
    k_gemm_bf16<<<gq,  256, 0, stream>>>(hid_b, wq_t, q_f, 2048, 4096, 4096);
    k_gemm_bf16<<<gkv, 256, 0, stream>>>(hid_b, wk_t, k_f, 2048, 1024, 4096);
    k_gemm_bf16<<<gkv, 256, 0, stream>>>(hid_b, wv_t, v_f, 2048, 1024, 4096);
    // 3) RoPE + layout transforms
    k_rope<<<(2048 * 4096) / 256, 256, 0, stream>>>(q_f, q_b, sinT, cosT, pos, NHq);
    k_rope<<<(2048 * 1024) / 256, 256, 0, stream>>>(k_f, k_b, sinT, cosT, pos, NKVh);
    k_vtrans<<<(2048 * 1024) / 256, 256, 0, stream>>>(v_f, vT_b);
    // 4) flash attention
    dim3 ga(NHq, 2048 / 128);
    k_attention<<<ga, 256, 0, stream>>>(q_b, k_b, vT_b, amask, attn_b);
    // 5) output projection -> f32 out
    k_gemm_bf16<<<gq, 256, 0, stream>>>(attn_b, wo_t, out, 2048, 4096, 4096);
}
